// CosFaceHeader_72997264163312
// MI455X (gfx1250) — compile-verified, matrix operand order checked by reading, not run on hardware
//
#include <hip/hip_runtime.h>
#include <math.h>

typedef __attribute__((ext_vector_type(16))) __bf16 v16bf;
typedef __attribute__((ext_vector_type(8)))  float  v8f;
typedef __attribute__((ext_vector_type(4)))  int    v4i;

#define B_SZ 512
#define D_SZ 512
#define SCALE_F 64.0f
#define MARGIN_F 0.35f
#define CT 64      // C-tile per block
#define BT 256     // B-tile per block
#define KB 32      // K-step (bf16 WMMA native K)
#define APITCH 40  // padded LDS pitch (bf16 elems) to dodge bank conflicts

#if defined(__HIP_DEVICE_COMPILE__) && __has_builtin(__builtin_amdgcn_global_load_async_to_lds_b128)
#define HAVE_ASYNC_LDS 1
#else
#define HAVE_ASYNC_LDS 0
#endif

#if HAVE_ASYNC_LDS
typedef __attribute__((address_space(1))) v4i g_v4i;  // global int4-vector
typedef __attribute__((address_space(3))) v4i l_v4i;  // LDS int4-vector
__device__ __forceinline__ g_v4i* glb_cast(const void* p) {
  return (g_v4i*)(unsigned long long)p;
}
__device__ __forceinline__ l_v4i* lds_cast(void* p) {
  // flat LDS address: low 32 bits are the LDS byte offset (aperture rule)
  return (l_v4i*)(unsigned int)(unsigned long long)p;
}
#endif

__device__ __forceinline__ void wait_async0() {
#if defined(__HIP_DEVICE_COMPILE__)
#if __has_builtin(__builtin_amdgcn_s_wait_asynccnt)
  __builtin_amdgcn_s_wait_asynccnt(0);
#else
  asm volatile("s_wait_asynccnt 0" ::: "memory");
#endif
#endif
}

__device__ __forceinline__ void splitBf(float v, __bf16& h, __bf16& l) {
  h = (__bf16)v;                 // RNE to bf16 (8-bit mantissa)
  l = (__bf16)(v - (float)h);    // residual, |l| <= 2^-9 |v|
}

// ---------------------------------------------------------------------------
// Kernel 1: L2-normalize x rows, emit bf16 hi/lo pair into workspace.
// ---------------------------------------------------------------------------
__global__ void xprep_kernel(const float* __restrict__ x,
                             __bf16* __restrict__ xh, __bf16* __restrict__ xl) {
  __shared__ float red[128];
  const int row = blockIdx.x;
  const int t = threadIdx.x;                 // 128 threads, 4 floats each
  float4 v = ((const float4*)x)[row * (D_SZ / 4) + t];
  red[t] = v.x * v.x + v.y * v.y + v.z * v.z + v.w * v.w;
  __syncthreads();
  for (int s = 64; s > 0; s >>= 1) {
    if (t < s) red[t] += red[t + s];
    __syncthreads();
  }
  const float inv = rsqrtf(fmaxf(red[0], 1e-24f));  // ~= 1/max(||x||, eps)
  float vv[4] = {v.x, v.y, v.z, v.w};
#pragma unroll
  for (int i = 0; i < 4; ++i) {
    __bf16 h, l;
    splitBf(vv[i] * inv, h, l);
    xh[row * D_SZ + t * 4 + i] = h;
    xl[row * D_SZ + t * 4 + i] = l;
  }
}

// ---------------------------------------------------------------------------
// Kernel 2: fused GEMM (bf16 WMMA, hi/lo 3-product fp32 emulation) +
//           on-the-fly wt row norms + margin + shifted-exp partial softmax.
// Double-buffered LDS; A tiles staged with async global->LDS copies.
// Grid: (ceil(C/CT), B/BT), 256 threads = 8 waves.
// ---------------------------------------------------------------------------
__global__ __launch_bounds__(256) void cosface_main(
    const float* __restrict__ wt, const __bf16* __restrict__ xh,
    const __bf16* __restrict__ xl, const int* __restrict__ gt,
    float* __restrict__ target_logit, float* __restrict__ partial,
    int C, int KSTEPS) {
  __shared__ __bf16 Ah[2][BT][APITCH];
  __shared__ __bf16 Al[2][BT][APITCH];
  __shared__ __bf16 Bh[2][CT][APITCH];
  __shared__ __bf16 Bl[2][CT][APITCH];
  __shared__ float cssq[CT][4];
  __shared__ float invNormW[CT];
  __shared__ float rowExp[BT];

  const int tid = threadIdx.x;
  const int c0 = blockIdx.x * CT;
  const int b0 = blockIdx.y * BT;

  rowExp[tid] = 0.0f;

  // staging roles: 4 threads per wt column, each owns one K-quarter
  const int colS = tid >> 2;      // 0..63
  const int kq = tid & 3;         // 0..3  (8 floats each)
  const int cG = c0 + colS;
  float ssq = 0.0f;

  // compute roles
  const int lane = tid & 31;
  const int wave = tid >> 5;
  const int wm = wave >> 1;       // 0..3 -> 64 rows each
  const int wn = wave & 1;        // 0..1 -> 32 cols each
  const int l15 = lane & 15;
  const int koff = (lane >> 4) * 8;  // per-lane K-half offset (ISA A/B striping)

  v8f acc[4][2];
#pragma unroll
  for (int mi = 0; mi < 4; ++mi)
#pragma unroll
    for (int ni = 0; ni < 2; ++ni) acc[mi][ni] = (v8f){0, 0, 0, 0, 0, 0, 0, 0};

  const float4* wt4 = (const float4*)wt;
  const uint4* xh4 = (const uint4*)xh;  // 8 bf16 per uint4
  const uint4* xl4 = (const uint4*)xl;

  // ---- staging helpers (lambdas keep shared-array access simple) ----
  auto stageA = [&](int buf, int k0) {
    const int base = ((b0 + tid) * D_SZ + k0) >> 3;  // uint4 = 8 bf16
#if HAVE_ASYNC_LDS
#pragma unroll
    for (int i = 0; i < 4; ++i) {
      __builtin_amdgcn_global_load_async_to_lds_b128(
          glb_cast(&xh4[base + i]), lds_cast(&Ah[buf][tid][i * 8]), 0, 0);
      __builtin_amdgcn_global_load_async_to_lds_b128(
          glb_cast(&xl4[base + i]), lds_cast(&Al[buf][tid][i * 8]), 0, 0);
    }
#else
#pragma unroll
    for (int i = 0; i < 4; ++i) {
      *(uint4*)&Ah[buf][tid][i * 8] = xh4[base + i];
      *(uint4*)&Al[buf][tid][i * 8] = xl4[base + i];
    }
#endif
  };

  auto loadB = [&](int k0, float4& f0, float4& f1) {
    f0 = make_float4(0.f, 0.f, 0.f, 0.f);
    f1 = f0;
    if (cG < C) {
      const int base = (cG * D_SZ + k0 + kq * 8) >> 2;
      f0 = wt4[base];
      f1 = wt4[base + 1];
      if (k0 + KB < D_SZ)  // gfx1250 global_prefetch_b8 of the slab after next
        __builtin_prefetch(&wt[cG * D_SZ + k0 + KB + kq * 8], 0, 3);
    }
  };

  auto storeB = [&](int buf, float4 f0, float4 f1) {
    float fv[8] = {f0.x, f0.y, f0.z, f0.w, f1.x, f1.y, f1.z, f1.w};
    union { __bf16 h[8]; uint4 q; } ph, pl;
#pragma unroll
    for (int i = 0; i < 8; ++i) {
      const float v = fv[i];
      ssq += v * v;
      splitBf(v, ph.h[i], pl.h[i]);
    }
    *(uint4*)&Bh[buf][colS][kq * 8] = ph.q;
    *(uint4*)&Bl[buf][colS][kq * 8] = pl.q;
  };

  // ---- prologue: fill buffer 0 ----
  {
    float4 f0, f1;
    loadB(0, f0, f1);
    stageA(0, 0);
    storeB(0, f0, f1);
    wait_async0();
    __syncthreads();
  }

  // ---- main pipeline: compute buf (kb&1) while staging buf ((kb+1)&1) ----
  for (int kb = 0; kb < KSTEPS; ++kb) {
    const int cur = kb & 1;
    const bool more = (kb + 1 < KSTEPS);
    float4 f0, f1;
    if (more) {
      loadB((kb + 1) * KB, f0, f1);    // global fp32 loads in flight
      stageA(cur ^ 1, (kb + 1) * KB);  // async copies into next buffer
    }

    // fragments per documented 16-bit 16x32 lane striping
    union Frag { v16bf v; uint4 q[2]; };
    Frag a_h[4], a_l[4], b_h[2], b_l[2];
#pragma unroll
    for (int mi = 0; mi < 4; ++mi) {
      const int r = wm * 64 + mi * 16 + l15;
      a_h[mi].q[0] = *(const uint4*)&Ah[cur][r][koff];
      a_h[mi].q[1] = *(const uint4*)&Ah[cur][r][16 + koff];
      a_l[mi].q[0] = *(const uint4*)&Al[cur][r][koff];
      a_l[mi].q[1] = *(const uint4*)&Al[cur][r][16 + koff];
    }
#pragma unroll
    for (int ni = 0; ni < 2; ++ni) {
      const int cL = wn * 32 + ni * 16 + l15;
      b_h[ni].q[0] = *(const uint4*)&Bh[cur][cL][koff];
      b_h[ni].q[1] = *(const uint4*)&Bh[cur][cL][16 + koff];
      b_l[ni].q[0] = *(const uint4*)&Bl[cur][cL][koff];
      b_l[ni].q[1] = *(const uint4*)&Bl[cur][cL][16 + koff];
    }
#pragma unroll
    for (int mi = 0; mi < 4; ++mi)
#pragma unroll
      for (int ni = 0; ni < 2; ++ni) {
        acc[mi][ni] = __builtin_amdgcn_wmma_f32_16x16x32_bf16(
            false, a_h[mi].v, false, b_h[ni].v, (short)0, acc[mi][ni], false, false);
        acc[mi][ni] = __builtin_amdgcn_wmma_f32_16x16x32_bf16(
            false, a_h[mi].v, false, b_l[ni].v, (short)0, acc[mi][ni], false, false);
        acc[mi][ni] = __builtin_amdgcn_wmma_f32_16x16x32_bf16(
            false, a_l[mi].v, false, b_h[ni].v, (short)0, acc[mi][ni], false, false);
      }

    if (more) storeB(cur ^ 1, f0, f1);  // convert + ds_store next B tile
    wait_async0();                      // next-buffer async copies landed
    __syncthreads();                    // everyone done reading cur / writing nxt
  }

  // ---- wt inverse norms (fp32, exact) ----
  cssq[colS][kq] = ssq;
  __syncthreads();
  if (tid < CT) {
    const float s = cssq[tid][0] + cssq[tid][1] + cssq[tid][2] + cssq[tid][3];
    invNormW[tid] = (s > 0.0f) ? rsqrtf(s) : 0.0f;
  }
  __syncthreads();

  // ---- epilogue: scale to logits, margin at gt, shifted-exp partial sums ----
#pragma unroll
  for (int mi = 0; mi < 4; ++mi) {
#pragma unroll
    for (int j = 0; j < 8; ++j) {
      const int rL = wm * 64 + mi * 16 + koff + j;  // C-layout: M = j + 8*(lane>=16)
      const int rG = b0 + rL;
      const int g = gt[rG];
      float esum = 0.0f;
#pragma unroll
      for (int ni = 0; ni < 2; ++ni) {
        const int cL = wn * 32 + ni * 16 + l15;     // C-layout: N = lane & 15
        const int cGl = c0 + cL;
        float logit = acc[mi][ni][j] * invNormW[cL] * SCALE_F;
        if (cGl == g) {                   // unique (b, gt) writer in whole grid
          logit -= SCALE_F * MARGIN_F;
          target_logit[rG] = logit;
        }
        esum += (cGl < C) ? __expf(logit - SCALE_F) : 0.0f;  // shift=64 >= max logit
      }
      // reduce across the 16 lanes sharing this row
      esum += __shfl_xor(esum, 1);
      esum += __shfl_xor(esum, 2);
      esum += __shfl_xor(esum, 4);
      esum += __shfl_xor(esum, 8);
      if (l15 == 0) atomicAdd(&rowExp[rL], esum);  // ds_add_f32
    }
  }
  __syncthreads();
  partial[blockIdx.x * B_SZ + b0 + tid] = rowExp[tid];
}

// ---------------------------------------------------------------------------
// Kernel 3: per-row logsumexp finish -> nll[b]
// ---------------------------------------------------------------------------
__global__ void finalize1(const float* __restrict__ partial,
                          const float* __restrict__ target_logit,
                          float* __restrict__ nll, int GC) {
  __shared__ float red[256];
  const int b = blockIdx.x, t = threadIdx.x;
  float s = 0.0f;
  for (int g = t; g < GC; g += 256) s += partial[g * B_SZ + b];
  red[t] = s;
  __syncthreads();
  for (int k = 128; k > 0; k >>= 1) {
    if (t < k) red[t] += red[t + k];
    __syncthreads();
  }
  if (t == 0) nll[b] = SCALE_F + logf(red[0]) - target_logit[b];
}

// ---------------------------------------------------------------------------
// Kernel 4: mean over batch -> scalar out
// ---------------------------------------------------------------------------
__global__ void finalize2(const float* __restrict__ nll, float* __restrict__ out) {
  __shared__ float red[256];
  const int t = threadIdx.x;
  red[t] = nll[t] + nll[t + 256];
  __syncthreads();
  for (int k = 128; k > 0; k >>= 1) {
    if (t < k) red[t] += red[t + k];
    __syncthreads();
  }
  if (t == 0) out[0] = red[0] * (1.0f / (float)B_SZ);
}

// ---------------------------------------------------------------------------
extern "C" void kernel_launch(void* const* d_in, const int* in_sizes, int n_in,
                              void* d_out, int out_size, void* d_ws, size_t ws_size,
                              hipStream_t stream) {
  (void)n_in; (void)out_size; (void)ws_size;
  const float* x = (const float*)d_in[0];
  const int* gt = (const int*)d_in[1];
  const float* wt = (const float*)d_in[2];
  const int C = in_sizes[2] / D_SZ;          // 100000
  const int GC = (C + CT - 1) / CT;          // 1563
  const int KSTEPS = D_SZ / KB;              // 16

  char* ws = (char*)d_ws;
  __bf16* xh = (__bf16*)(ws);                                   // 512 KB
  __bf16* xl = (__bf16*)(ws + 524288);                          // 512 KB
  float* target_logit = (float*)(ws + 1048576);                 // 2 KB (pad 4 KB)
  float* partial = (float*)(ws + 1052672);                      // GC*512*4 B
  float* nll = (float*)(ws + 1052672 + (size_t)GC * B_SZ * 4);  // 2 KB

  xprep_kernel<<<B_SZ, 128, 0, stream>>>(x, xh, xl);
  dim3 grid(GC, B_SZ / BT);
  cosface_main<<<grid, 256, 0, stream>>>(wt, xh, xl, gt, target_logit, partial,
                                         C, KSTEPS);
  finalize1<<<B_SZ, 256, 0, stream>>>(partial, target_logit, nll, GC);
  finalize2<<<1, 256, 0, stream>>>(nll, (float*)d_out);
}
